// CrossAttentionBlock_39556648796229
// MI455X (gfx1250) — matrix operand order repaired and checked by reading, compiled, not run on hardware
//
#include <hip/hip_runtime.h>

typedef __attribute__((ext_vector_type(16))) _Float16 v16h;
typedef __attribute__((ext_vector_type(8)))  _Float16 v8h;
typedef __attribute__((ext_vector_type(8)))  float    v8f;

#define EPSV 1e-3f

#define WMMA_F16(A, B, C) \
    __builtin_amdgcn_wmma_f32_16x16x32_f16(false, (A), false, (B), (short)0, (C), false, false)

// ---------------------------------------------------------------------------
// Small prep kernels
// ---------------------------------------------------------------------------

// Fold BatchNorm (inference) into the pin projection:
//   s[c] = bn_g[c] * rsqrt(bn_v[c]+eps);  t[c] = bn_b[c] - bn_m[c]*s[c]
//   bias'[u] = pin_b[u] + sum_c t[c]*pin_w[c][u]
__global__ void bn_fold_kernel(const float* __restrict__ pin_w,
                               const float* __restrict__ pin_b,
                               const float* __restrict__ bn_g,
                               const float* __restrict__ bn_b,
                               const float* __restrict__ bn_m,
                               const float* __restrict__ bn_v,
                               float* __restrict__ s_out,
                               float* __restrict__ bias_out) {
    int u = threadIdx.x;  // 256 threads
    float su = bn_g[u] * rsqrtf(bn_v[u] + EPSV);
    s_out[u] = su;
    float acc = 0.0f;
    for (int c = 0; c < 256; ++c) {
        float sc = bn_g[c] * rsqrtf(bn_v[c] + EPSV);
        float tc = bn_b[c] - bn_m[c] * sc;
        acc += tc * pin_w[c * 256 + u];
    }
    bias_out[u] = pin_b[u] + acc;
}

// Transpose weight [K][N] f32 -> [N][K] f16, optional per-K scale (BN fold).
__global__ void weightT_f16_kernel(const float* __restrict__ w,
                                   _Float16* __restrict__ wt,
                                   int K, int N, const float* __restrict__ scale) {
    int idx = blockIdx.x * blockDim.x + threadIdx.x;
    if (idx >= N * K) return;
    int n = idx / K, c = idx % K;
    float v = w[(size_t)c * N + n];
    if (scale) v *= scale[c];
    wt[idx] = (_Float16)v;
}

__global__ void cvt_f16_kernel(const float* __restrict__ in,
                               _Float16* __restrict__ out, int n) {
    int i = blockIdx.x * blockDim.x + threadIdx.x;
    if (i < n) out[i] = (_Float16)in[i];
}

__global__ void copy_f32_kernel(const float* __restrict__ in,
                                float* __restrict__ out, int n) {
    int i = blockIdx.x * blockDim.x + threadIdx.x;
    if (i < n) out[i] = in[i];
}

// Fused 3x LayerNorm over the 256-wide channel dim; one block per row.
__global__ void layernorm3_kernel(const float* __restrict__ h,
                                  const float* __restrict__ g1, const float* __restrict__ b1,
                                  const float* __restrict__ g2, const float* __restrict__ b2,
                                  const float* __restrict__ g3, const float* __restrict__ b3,
                                  _Float16* __restrict__ o1,
                                  _Float16* __restrict__ o2,
                                  _Float16* __restrict__ o3) {
    __shared__ float red[256];
    int row = blockIdx.x, t = threadIdx.x;
    float v = h[(size_t)row * 256 + t];
    red[t] = v;
    __syncthreads();
    for (int s = 128; s > 0; s >>= 1) { if (t < s) red[t] += red[t + s]; __syncthreads(); }
    float mu = red[0] * (1.0f / 256.0f);
    __syncthreads();
    float d = v - mu;
    red[t] = d * d;
    __syncthreads();
    for (int s = 128; s > 0; s >>= 1) { if (t < s) red[t] += red[t + s]; __syncthreads(); }
    float rs = rsqrtf(red[0] * (1.0f / 256.0f) + EPSV);
    float nv = d * rs;
    size_t idx = (size_t)row * 256 + t;
    o1[idx] = (_Float16)(nv * g1[t] + b1[t]);
    o2[idx] = (_Float16)(nv * g2[t] + b2[t]);
    o3[idx] = (_Float16)(nv * g3[t] + b3[t]);
}

// ---------------------------------------------------------------------------
// Register-blocked WMMA GEMM: D = A[MxK](f16) * WT[N][K](f16) + bias.
// One wave computes a 32x64 tile: 2 A-frags x 4 B-frags -> 8 WMMAs per
// 32-deep K-step (48 B loaded per WMMA vs 128 B unblocked).
// flags: bit0 = ReLU, bit1 = accumulate into outf32
// ---------------------------------------------------------------------------
__global__ void __launch_bounds__(256)
gemm_wmma_kernel(const _Float16* __restrict__ A, const _Float16* __restrict__ WT,
                 const float* __restrict__ bias,
                 float* __restrict__ outf32, _Float16* __restrict__ outf16,
                 const float* __restrict__ residual,
                 int M, int N, int K, int flags) {
    const int lane = threadIdx.x & 31;
    const int wave = blockIdx.x * (blockDim.x >> 5) + (threadIdx.x >> 5);
    const int ngrp = N >> 6;                 // 64-column groups
    const int mt = wave / ngrp;              // 32-row tile index
    const int ng = wave - mt * ngrp;
    if (mt >= (M >> 5)) return;              // wave-uniform guard

    const int m0 = mt << 5, n0 = ng << 6;
    v8f acc[2][4] = {};
    for (int kk = 0; kk < K; kk += 32) {
        v16h a[2];
#pragma unroll
        for (int i = 0; i < 2; ++i) {
            // A fragment: m = lane%16, k = (lane/16)*16 + e
            const _Float16* ap = A + (size_t)(m0 + i * 16 + (lane & 15)) * K + kk +
                                 ((lane >> 4) << 4);
            v8h lo = *(const v8h*)ap;
            v8h hi = *(const v8h*)(ap + 8);
#pragma unroll
            for (int e = 0; e < 8; ++e) { a[i][e] = lo[e]; a[i][e + 8] = hi[e]; }
        }
#pragma unroll
        for (int j = 0; j < 4; ++j) {
            // B fragment from transposed weight: n = lane%16, k = (lane/16)*16 + e
            v16h b = *(const v16h*)(WT + (size_t)(n0 + j * 16 + (lane & 15)) * K + kk +
                                    ((lane >> 4) << 4));
            acc[0][j] = WMMA_F16(a[0], b, acc[0][j]);
            acc[1][j] = WMMA_F16(a[1], b, acc[1][j]);
        }
    }
#pragma unroll
    for (int i = 0; i < 2; ++i) {
#pragma unroll
        for (int j = 0; j < 4; ++j) {
            int n = n0 + j * 16 + (lane & 15);
            float bb = bias ? bias[n] : 0.0f;
#pragma unroll
            for (int r = 0; r < 8; ++r) {
                int m = m0 + i * 16 + ((lane >> 4) << 3) + r;
                float v = acc[i][j][r] + bb;
                if (flags & 1) v = fmaxf(v, 0.0f);
                size_t idx = (size_t)m * N + n;
                if (residual) v += residual[idx];
                if (outf32) { if (flags & 2) outf32[idx] += v; else outf32[idx] = v; }
                if (outf16) outf16[idx] = (_Float16)v;
            }
        }
    }
}

// ---------------------------------------------------------------------------
// Flash attention: one wave per (batch, head, 16-query tile).
// Per 32-key chunk: 2 score WMMAs (K = head_dim = 32), online softmax with
// shfl row reductions, P transposed via LDS, 2 P*V WMMAs (K = 32 keys).
// Accumulates (out += attn) into the f32 sum buffer, head columns disjoint.
// ---------------------------------------------------------------------------
__global__ void __launch_bounds__(32)
attn_flash_kernel(const _Float16* __restrict__ q, const _Float16* __restrict__ k,
                  const _Float16* __restrict__ v, float* __restrict__ out,
                  int L, int Lk) {
    __shared__ _Float16 P[16 * 32];
    const int lane = threadIdx.x & 31;
    const int qtiles = L >> 4;
    const int qt = blockIdx.x % qtiles;
    const int hb = blockIdx.x / qtiles;
    const int head = hb & 7;
    const int b = hb >> 3;
    const int col0 = head * 32;
    const int qrow0 = b * L + (qt << 4);

    // Q fragment (A layout, K = head_dim = 32)
    const _Float16* qp = q + (size_t)(qrow0 + (lane & 15)) * 256 + col0 + ((lane >> 4) << 4);
    v8h qlo = *(const v8h*)qp;
    v8h qhi = *(const v8h*)(qp + 8);
    v16h qa;
#pragma unroll
    for (int i = 0; i < 8; ++i) { qa[i] = qlo[i]; qa[i + 8] = qhi[i]; }

    v8f o0 = {}, o1 = {};
    float rmax[8], rsum[8];
#pragma unroll
    for (int r = 0; r < 8; ++r) { rmax[r] = -3.0e38f; rsum[r] = 0.0f; }
    const float sc = 0.0625f;  // UNITS^-0.5

    for (int kb = 0; kb < Lk; kb += 32) {
        // K fragments (B layout): n = key (lane%16), k = channel (lane/16)*16+e
        const _Float16* kp = k + (size_t)(b * Lk + kb + (lane & 15)) * 256 + col0 +
                             ((lane >> 4) << 4);
        v16h kf0 = *(const v16h*)kp;
        v16h kf1 = *(const v16h*)(kp + (size_t)16 * 256);
        v8f s0 = {}, s1 = {};
        s0 = WMMA_F16(qa, kf0, s0);
        s1 = WMMA_F16(qa, kf1, s1);
#pragma unroll
        for (int r = 0; r < 8; ++r) {
            float a0 = s0[r] * sc, a1 = s1[r] * sc;
            float mx = fmaxf(a0, a1);
#pragma unroll
            for (int o = 8; o > 0; o >>= 1) mx = fmaxf(mx, __shfl_xor(mx, o, 32));
            float nm = fmaxf(rmax[r], mx);
            float alpha = __expf(rmax[r] - nm);
            float p0 = __expf(a0 - nm), p1 = __expf(a1 - nm);
            float ps = p0 + p1;
#pragma unroll
            for (int o = 8; o > 0; o >>= 1) ps += __shfl_xor(ps, o, 32);
            rsum[r] = rsum[r] * alpha + ps;
            rmax[r] = nm;
            o0[r] *= alpha; o1[r] *= alpha;
            int m = ((lane >> 4) << 3) + r;  // D layout row
            P[m * 32 + (lane & 15)]      = (_Float16)p0;
            P[m * 32 + 16 + (lane & 15)] = (_Float16)p1;
        }
        asm volatile("s_wait_dscnt 0x0" ::: "memory");
        // P fragment (A layout, K = 32 keys)
        const _Float16* pp = &P[(lane & 15) * 32 + ((lane >> 4) << 4)];
        v8h plo = *(const v8h*)pp;
        v8h phi = *(const v8h*)(pp + 8);
        v16h pa;
#pragma unroll
        for (int i = 0; i < 8; ++i) { pa[i] = plo[i]; pa[i + 8] = phi[i]; }
        // V fragments (B layout): n = out channel (lane%16), k = key, strided
        const _Float16* vp = v + (size_t)(b * Lk + kb + ((lane >> 4) << 4)) * 256 + col0 +
                             (lane & 15);
        v16h vf0, vf1;
#pragma unroll
        for (int e = 0; e < 16; ++e) {
            vf0[e] = vp[(size_t)e * 256];
            vf1[e] = vp[(size_t)e * 256 + 16];
        }
        o0 = WMMA_F16(pa, vf0, o0);
        o1 = WMMA_F16(pa, vf1, o1);
    }
#pragma unroll
    for (int r = 0; r < 8; ++r) {
        float inv = 1.0f / rsum[r];
        int m = qrow0 + ((lane >> 4) << 3) + r;
        out[(size_t)m * 256 + col0 + (lane & 15)]      += o0[r] * inv;
        out[(size_t)m * 256 + col0 + 16 + (lane & 15)] += o1[r] * inv;
    }
}

// ---------------------------------------------------------------------------
// Host launch
// ---------------------------------------------------------------------------
extern "C" void kernel_launch(void* const* d_in, const int* in_sizes, int n_in,
                              void* d_out, int out_size, void* d_ws, size_t ws_size,
                              hipStream_t stream) {
    const int S = 12, Bb = 2;
    const int L = S * S * S;      // 1728
    const int M = Bb * L;         // 3456
    const int C = 256, U = 256, F = 1024;

    const float* x     = (const float*)d_in[0];
    const float* ctx   = (const float*)d_in[1];
    const float* bn_g  = (const float*)d_in[2];
    const float* bn_b  = (const float*)d_in[3];
    const float* bn_m  = (const float*)d_in[4];
    const float* bn_v  = (const float*)d_in[5];
    const float* ln1_g = (const float*)d_in[6];
    const float* ln1_b = (const float*)d_in[7];
    const float* ln2_g = (const float*)d_in[8];
    const float* ln2_b = (const float*)d_in[9];
    const float* ln3_g = (const float*)d_in[10];
    const float* ln3_b = (const float*)d_in[11];
    const float* pin_w = (const float*)d_in[12];
    const float* pin_b = (const float*)d_in[13];
    const float* q_w   = (const float*)d_in[14];
    const float* q_b   = (const float*)d_in[15];
    const float* k_w   = (const float*)d_in[16];
    const float* k_b   = (const float*)d_in[17];
    const float* v_w   = (const float*)d_in[18];
    const float* v_b   = (const float*)d_in[19];
    const float* ff1_w = (const float*)d_in[20];
    const float* ff1_b = (const float*)d_in[21];
    const float* ff2_w = (const float*)d_in[22];
    const float* ff2_b = (const float*)d_in[23];
    const float* pout_w= (const float*)d_in[24];
    const float* pout_b= (const float*)d_in[25];

    size_t off = 0;
    auto alloc = [&](size_t bytes) -> void* {
        void* p = (char*)d_ws + off;
        off += (bytes + 255) & ~(size_t)255;
        return p;
    };

    float*     s_buf = (float*)alloc(C * 4);
    float*     binp  = (float*)alloc(U * 4);
    _Float16*  pinT  = (_Float16*)alloc((size_t)C * U * 2);
    _Float16*  qT    = (_Float16*)alloc((size_t)U * U * 2);
    _Float16*  kT    = (_Float16*)alloc((size_t)U * U * 2);
    _Float16*  vT    = (_Float16*)alloc((size_t)U * U * 2);
    _Float16*  ff1T  = (_Float16*)alloc((size_t)U * F * 2);
    _Float16*  ff2T  = (_Float16*)alloc((size_t)F * U * 2);
    _Float16*  poutT = (_Float16*)alloc((size_t)U * U * 2);
    _Float16*  xh    = (_Float16*)alloc((size_t)M * C * 2);
    _Float16*  ch    = (_Float16*)alloc((size_t)M * C * 2);
    float*     h32   = (float*)alloc((size_t)M * U * 4);
    _Float16*  h16   = (_Float16*)alloc((size_t)M * U * 2);
    _Float16*  ln1h  = (_Float16*)alloc((size_t)M * U * 2);
    _Float16*  ln2h  = (_Float16*)alloc((size_t)M * U * 2);
    _Float16*  ln3h  = (_Float16*)alloc((size_t)M * U * 2);
    _Float16*  q1    = (_Float16*)alloc((size_t)M * U * 2);
    _Float16*  k1    = (_Float16*)alloc((size_t)M * U * 2);
    _Float16*  v1    = (_Float16*)alloc((size_t)M * U * 2);
    _Float16*  q2    = (_Float16*)alloc((size_t)M * U * 2);
    _Float16*  k2    = (_Float16*)alloc((size_t)M * U * 2);
    _Float16*  v2    = (_Float16*)alloc((size_t)M * U * 2);
    _Float16*  ff1h  = (_Float16*)alloc((size_t)M * F * 2);
    float*     ot32  = (float*)alloc((size_t)M * U * 4);
    _Float16*  ot16  = (_Float16*)alloc((size_t)M * U * 2);

    auto cdiv = [](int a, int b) { return (a + b - 1) / b; };
    // 32x64 tile per wave, 8 waves per block
    auto gemm_grid = [&](int Mm, int Nn) { return (Mm / 32) * (Nn / 64) / 8; };

    // 1) BN fold + weight prep (f16, transposed)
    bn_fold_kernel<<<1, 256, 0, stream>>>(pin_w, pin_b, bn_g, bn_b, bn_m, bn_v, s_buf, binp);
    weightT_f16_kernel<<<cdiv(C * U, 256), 256, 0, stream>>>(pin_w, pinT, C, U, s_buf);
    weightT_f16_kernel<<<cdiv(U * U, 256), 256, 0, stream>>>(q_w, qT, U, U, nullptr);
    weightT_f16_kernel<<<cdiv(U * U, 256), 256, 0, stream>>>(k_w, kT, U, U, nullptr);
    weightT_f16_kernel<<<cdiv(U * U, 256), 256, 0, stream>>>(v_w, vT, U, U, nullptr);
    weightT_f16_kernel<<<cdiv(U * F, 256), 256, 0, stream>>>(ff1_w, ff1T, U, F, nullptr);
    weightT_f16_kernel<<<cdiv(F * U, 256), 256, 0, stream>>>(ff2_w, ff2T, F, U, nullptr);
    weightT_f16_kernel<<<cdiv(U * U, 256), 256, 0, stream>>>(pout_w, poutT, U, U, nullptr);

    // 2) Activations to f16
    cvt_f16_kernel<<<cdiv(M * C, 256), 256, 0, stream>>>(x, xh, M * C);
    cvt_f16_kernel<<<cdiv(M * C, 256), 256, 0, stream>>>(ctx, ch, M * C);

    // 3) h = relu(bn(x) @ pin_w + pin_b)  (BN folded into pinT/binp)
    gemm_wmma_kernel<<<gemm_grid(M, U), 256, 0, stream>>>(xh, pinT, binp, h32, h16,
                                                          nullptr, M, U, C, 1);
    // out_total = h  (self-attention residual)
    copy_f32_kernel<<<cdiv(M * U, 256), 256, 0, stream>>>(h32, ot32, M * U);

    // 4) Three LayerNorms of h
    layernorm3_kernel<<<M, 256, 0, stream>>>(h32, ln1_g, ln1_b, ln2_g, ln2_b, ln3_g, ln3_b,
                                             ln1h, ln2h, ln3h);

    // 5) Projections
    gemm_wmma_kernel<<<gemm_grid(M, U), 256, 0, stream>>>(ln1h, qT, q_b, nullptr, q1,
                                                          nullptr, M, U, U, 0);
    gemm_wmma_kernel<<<gemm_grid(M, U), 256, 0, stream>>>(h16,  kT, k_b, nullptr, k1,
                                                          nullptr, M, U, U, 0);
    gemm_wmma_kernel<<<gemm_grid(M, U), 256, 0, stream>>>(h16,  vT, v_b, nullptr, v1,
                                                          nullptr, M, U, U, 0);
    gemm_wmma_kernel<<<gemm_grid(M, U), 256, 0, stream>>>(ln2h, qT, q_b, nullptr, q2,
                                                          nullptr, M, U, U, 0);
    gemm_wmma_kernel<<<gemm_grid(M, U), 256, 0, stream>>>(ch,   kT, k_b, nullptr, k2,
                                                          nullptr, M, U, U, 0);
    gemm_wmma_kernel<<<gemm_grid(M, U), 256, 0, stream>>>(ch,   vT, v_b, nullptr, v2,
                                                          nullptr, M, U, U, 0);

    // 6) Self- and cross-attention, accumulate into out_total
    const int attn_blocks = Bb * 8 * (L / 16);
    attn_flash_kernel<<<attn_blocks, 32, 0, stream>>>(q1, k1, v1, ot32, L, L);
    attn_flash_kernel<<<attn_blocks, 32, 0, stream>>>(q2, k2, v2, ot32, L, L);

    // 7) FFN: relu(ln3h @ ff1 + b) @ ff2 + b, accumulated into out_total
    gemm_wmma_kernel<<<gemm_grid(M, F), 256, 0, stream>>>(ln3h, ff1T, ff1_b, nullptr, ff1h,
                                                          nullptr, M, F, U, 1);
    gemm_wmma_kernel<<<gemm_grid(M, U), 256, 0, stream>>>(ff1h, ff2T, ff2_b, ot32, nullptr,
                                                          nullptr, M, U, F, 2);

    // 8) Final: relu(out_total @ pout + b) + x
    cvt_f16_kernel<<<cdiv(M * U, 256), 256, 0, stream>>>(ot32, ot16, M * U);
    gemm_wmma_kernel<<<gemm_grid(M, U), 256, 0, stream>>>(ot16, poutT, pout_b,
                                                          (float*)d_out, nullptr, x,
                                                          M, U, U, 1);
}
